// CGCNN_21131239096637
// MI455X (gfx1250) — matrix operand order repaired and checked by reading, not compiled, not run on hardware
//
#include <hip/hip_runtime.h>
#include <hip/hip_bf16.h>
#include <math.h>

typedef __attribute__((ext_vector_type(16))) _Float16 v16h;
typedef __attribute__((ext_vector_type(8)))  _Float16 v8h;
typedef __attribute__((ext_vector_type(8)))  float    v8f;

#define N_NODES 50000
#define N_EDGES 800000
#define N_GRAPHS 1000
#define D_NODE 92
#define D_EDGE 41
#define D_HID 128
#define ZIN 297      // 2*D_HID + D_EDGE
#define ZPAD 320     // padded to 10 * 32
#define KSTEPS 10
#define FRAG_HALVES 512   // 32 lanes * 16 halves per B fragment
#define EPB 64            // edges per block (4 waves x 16-edge tile each)

__device__ __forceinline__ float leaky(float v) { return v > 0.f ? v : 0.01f * v; }

// ---------------------------------------------------------------------------
// Pack Wf1, Ws1, Wf2, Ws2 ([128][297] row-major, f32) into WMMA-B fragment
// order (f16, zero padded K->320).  B order used here (contiguous k, same as A): lane L holds
// column N = L%16; halves e=0..15 hold K = (L>>4)*16 + e within each K step.
// ---------------------------------------------------------------------------
__global__ void k_pack_B(const float* __restrict__ Wf1, const float* __restrict__ Ws1,
                         const float* __restrict__ Wf2, const float* __restrict__ Ws2,
                         _Float16* __restrict__ packB) {
    int idx = blockIdx.x * 256 + threadIdx.x;
    const int total = 4 * 8 * KSTEPS * FRAG_HALVES;
    if (idx >= total) return;
    int e    = idx & 15;
    int lane = (idx >> 4) & 31;
    int ks   = (idx >> 9) % KSTEPS;
    int nt   = (idx / (KSTEPS * FRAG_HALVES)) & 7;
    int m    = idx / (8 * KSTEPS * FRAG_HALVES);
    int N = nt * 16 + (lane & 15);
    int K = ks * 32 + (lane >> 4) * 16 + e;
    const float* W = (m == 0) ? Wf1 : (m == 1) ? Ws1 : (m == 2) ? Wf2 : Ws2;
    float v = (K < ZIN) ? W[(size_t)N * ZIN + K] : 0.f;
    packB[idx] = (_Float16)v;
}

// ---------------------------------------------------------------------------
// h0 = leaky_relu(x @ W_node^T + b_node); also emit f16 copy.
// ---------------------------------------------------------------------------
__global__ __launch_bounds__(128) void k_node_embed(const float* __restrict__ x,
                                                    const float* __restrict__ Wn,
                                                    const float* __restrict__ bn,
                                                    float* __restrict__ h,
                                                    _Float16* __restrict__ h16) {
    __shared__ float xs[D_NODE];
    int n = blockIdx.x, c = threadIdx.x;
    for (int i = c; i < D_NODE; i += 128) xs[i] = x[(size_t)n * D_NODE + i];
    __syncthreads();
    float acc = bn[c];
    const float* wr = Wn + (size_t)c * D_NODE;
    #pragma unroll 4
    for (int k = 0; k < D_NODE; ++k) acc = fmaf(xs[k], wr[k], acc);
    acc = leaky(acc);
    size_t o = (size_t)n * D_HID + c;
    h[o] = acc;
    h16[o] = (_Float16)acc;
}

// ---------------------------------------------------------------------------
// CGConv message + scatter-add.  Block = 128 threads (4 waves), 64 edges.
// Each wave owns a 16-edge tile and computes ALL 8 N-tiles (full 128 chans):
// K-step outer (1 A load from LDS), N-tile inner (16 WMMAs), 16 independent
// accumulator chains -> no WMMA->WMMA RAW hazards, 8x fewer ds loads/WMMA.
// ---------------------------------------------------------------------------
__global__ __launch_bounds__(128) void k_edge_msg(const _Float16* __restrict__ h16,
                                                  const float* __restrict__ edge_attr,
                                                  const int* __restrict__ eidx,
                                                  const _Float16* __restrict__ packB,
                                                  const float* __restrict__ bf,
                                                  const float* __restrict__ bs,
                                                  float* __restrict__ agg,
                                                  int matBase) {
    __shared__ __attribute__((aligned(16))) _Float16 Z[EPB][ZPAD];   // 40 KB
    __shared__ int sdst[EPB], ssrc[EPB];
    const int tid = threadIdx.x;
    const int e0  = blockIdx.x * EPB;
    const int* srcIdx = eidx;            // edge_index[0]
    const int* dstIdx = eidx + N_EDGES;  // edge_index[1]

    if (tid < EPB)            sdst[tid]       = dstIdx[e0 + tid];
    else if (tid < 2 * EPB)   ssrc[tid - EPB] = srcIdx[e0 + tid - EPB];
    __syncthreads();

    // Stage [h[dst] | h[src]] : EPB rows x 256 halves, as 16-byte vectors.
    for (int v = tid; v < EPB * 32; v += 128) {
        int r = v >> 5;
        int c = (v & 31) * 8;
        const _Float16* sp = (c < D_HID) ? (h16 + (size_t)sdst[r] * D_HID + c)
                                         : (h16 + (size_t)ssrc[r] * D_HID + (c - D_HID));
        *(v8h*)&Z[r][c] = *(const v8h*)sp;
    }
    // Stage edge_attr + zero pad : EPB rows x 64 halves.
    for (int v = tid; v < EPB * 64; v += 128) {
        int r = v >> 6;
        int c = 256 + (v & 63);
        Z[r][c] = (c < ZIN) ? (_Float16)edge_attr[(size_t)(e0 + r) * D_EDGE + (c - 256)]
                            : (_Float16)0.f;
    }
    __syncthreads();

    const int lane  = tid & 31;
    const int wt    = tid >> 5;                 // wave's 16-edge sub-tile 0..3
    const int row   = wt * 16 + (lane & 15);    // A row (edge within block)
    const int kbase = (lane >> 4) * 16;         // A K-sub-base (contiguous, matches packed B)

    v8f zero = {};
    v8f accF[8], accS[8];
    #pragma unroll
    for (int nt = 0; nt < 8; ++nt) { accF[nt] = zero; accS[nt] = zero; }

    const _Float16* pB = packB + (size_t)(matBase * 8) * KSTEPS * FRAG_HALVES + lane * 16;
    const size_t NT_STRIDE = (size_t)KSTEPS * FRAG_HALVES;      // between N-tiles
    const size_t S_OFF     = (size_t)8 * KSTEPS * FRAG_HALVES;  // F-mats -> S-mats

    for (int ks = 0; ks < KSTEPS; ++ks) {
        int k0 = ks * 32 + kbase;
        v8h c0 = *(const v8h*)&Z[row][k0];        // K = kbase .. kbase+7
        v8h c1 = *(const v8h*)&Z[row][k0 + 8];    // K = kbase+8 .. kbase+15
        v16h a;
        #pragma unroll
        for (int i = 0; i < 8; ++i) { a[i] = c0[i]; a[i + 8] = c1[i]; }
        const _Float16* pk = pB + (size_t)ks * FRAG_HALVES;
        #pragma unroll
        for (int nt = 0; nt < 8; ++nt) {
            v16h bF = *(const v16h*)(pk + (size_t)nt * NT_STRIDE);
            v16h bS = *(const v16h*)(pk + S_OFF + (size_t)nt * NT_STRIDE);
            accF[nt] = __builtin_amdgcn_wmma_f32_16x16x32_f16(false, a, false, bF, (short)0, accF[nt], false, false);
            accS[nt] = __builtin_amdgcn_wmma_f32_16x16x32_f16(false, a, false, bS, (short)0, accS[nt], false, false);
        }
    }

    // D layout: VGPR r -> edge row r (lanes 0-15) / r+8 (lanes 16-31); col = lane%16.
    const int col   = lane & 15;
    const int rbase = wt * 16 + (lane >> 4) * 8;
    #pragma unroll
    for (int nt = 0; nt < 8; ++nt) {
        const int ch = nt * 16 + col;
        const float biasF = bf[ch];
        const float biasS = bs[ch];
        #pragma unroll
        for (int r = 0; r < 8; ++r) {
            int el = rbase + r;
            float xf = accF[nt][r] + biasF;
            float xs = accS[nt][r] + biasS;
            float sg = 1.f / (1.f + __expf(-xf));
            float sp = (xs > 20.f) ? xs : ((xs < -20.f) ? __expf(xs) : log1pf(__expf(xs)));
            atomicAdd(&agg[(size_t)sdst[el] * D_HID + ch], sg * sp);
        }
    }
}

// ---------------------------------------------------------------------------
// h_out = leaky_relu(h_in + agg); optional f16 refresh.
// ---------------------------------------------------------------------------
__global__ void k_residual(const float* __restrict__ hin, const float* __restrict__ agg,
                           float* __restrict__ hout, _Float16* __restrict__ h16out, int n) {
    int i = blockIdx.x * 256 + threadIdx.x;
    if (i >= n) return;
    float v = leaky(hin[i] + agg[i]);
    hout[i] = v;
    if (h16out) h16out[i] = (_Float16)v;
}

// ---------------------------------------------------------------------------
// pooled[batch[n]] += atom_embs[n]
// ---------------------------------------------------------------------------
__global__ void k_pool(const float* __restrict__ atom, const int* __restrict__ batch,
                       float* __restrict__ pooled, int n) {
    int i = blockIdx.x * 256 + threadIdx.x;
    if (i >= n) return;
    int node = i >> 7, c = i & 127;
    atomicAdd(&pooled[(size_t)batch[node] * D_HID + c], atom[i]);
}

// ---------------------------------------------------------------------------
// Per-graph: L2-normalize, g = leaky(g@W1^T+b1), out = g@W2^T+b2.
// ---------------------------------------------------------------------------
__global__ __launch_bounds__(128) void k_head(const float* __restrict__ pooled,
                                              const float* __restrict__ W1, const float* __restrict__ b1,
                                              const float* __restrict__ W2, const float* __restrict__ b2,
                                              float* __restrict__ out) {
    __shared__ float gn[D_HID];
    __shared__ float red[D_HID];
    int g = blockIdx.x, t = threadIdx.x;
    float v = pooled[(size_t)g * D_HID + t];
    red[t] = v * v;
    __syncthreads();
    for (int s = 64; s > 0; s >>= 1) { if (t < s) red[t] += red[t + s]; __syncthreads(); }
    float norm = fmaxf(sqrtf(red[0]), 1e-12f);
    gn[t] = v / norm;
    __syncthreads();
    float acc = b1[t];
    const float* wr = W1 + (size_t)t * D_HID;
    #pragma unroll 4
    for (int k = 0; k < D_HID; ++k) acc = fmaf(gn[k], wr[k], acc);
    acc = leaky(acc);
    red[t] = acc * W2[t];
    __syncthreads();
    for (int s = 64; s > 0; s >>= 1) { if (t < s) red[t] += red[t + s]; __syncthreads(); }
    if (t == 0) out[g] = red[0] + b2[0];
}

extern "C" void kernel_launch(void* const* d_in, const int* in_sizes, int n_in,
                              void* d_out, int out_size, void* d_ws, size_t ws_size,
                              hipStream_t stream) {
    const float* x     = (const float*)d_in[0];
    const int*   eidx  = (const int*)d_in[1];
    const float* eattr = (const float*)d_in[2];
    const int*   batch = (const int*)d_in[3];
    const float* Wn  = (const float*)d_in[4];  const float* bn  = (const float*)d_in[5];
    const float* Wf1 = (const float*)d_in[6];  const float* bf1 = (const float*)d_in[7];
    const float* Ws1 = (const float*)d_in[8];  const float* bs1 = (const float*)d_in[9];
    const float* Wf2 = (const float*)d_in[10]; const float* bf2 = (const float*)d_in[11];
    const float* Ws2 = (const float*)d_in[12]; const float* bs2 = (const float*)d_in[13];
    const float* W1  = (const float*)d_in[14]; const float* b1  = (const float*)d_in[15];
    const float* W2  = (const float*)d_in[16]; const float* b2  = (const float*)d_in[17];

    float* out  = (float*)d_out;       // [1000]
    float* atom = out + N_GRAPHS;      // [50000*128]

    const size_t HN = (size_t)N_NODES * D_HID;
    char* ws = (char*)d_ws;
    float*    h      = (float*)ws;     ws += HN * 4;
    float*    agg    = (float*)ws;     ws += HN * 4;
    _Float16* h16    = (_Float16*)ws;  ws += HN * 2;
    _Float16* packB  = (_Float16*)ws;  ws += (size_t)4 * 8 * KSTEPS * FRAG_HALVES * 2;
    float*    pooled = (float*)ws;     ws += (size_t)N_GRAPHS * D_HID * 4;

    const int NBLK  = N_EDGES / EPB;            // 12500
    const int HN_I  = (int)HN;                  // 6,400,000
    const int HN_BLK = (HN_I + 255) / 256;

    k_pack_B<<<(4 * 8 * KSTEPS * FRAG_HALVES + 255) / 256, 256, 0, stream>>>(Wf1, Ws1, Wf2, Ws2, packB);
    k_node_embed<<<N_NODES, 128, 0, stream>>>(x, Wn, bn, h, h16);

    // Layer 1
    hipMemsetAsync(agg, 0, HN * 4, stream);
    k_edge_msg<<<NBLK, 128, 0, stream>>>(h16, eattr, eidx, packB, bf1, bs1, agg, 0);
    k_residual<<<HN_BLK, 256, 0, stream>>>(h, agg, h, h16, HN_I);

    // Layer 2 -> atom_embs (d_out tail)
    hipMemsetAsync(agg, 0, HN * 4, stream);
    k_edge_msg<<<NBLK, 128, 0, stream>>>(h16, eattr, eidx, packB, bf2, bs2, agg, 2);
    k_residual<<<HN_BLK, 256, 0, stream>>>(h, agg, atom, nullptr, HN_I);

    // Pool + head
    hipMemsetAsync(pooled, 0, (size_t)N_GRAPHS * D_HID * 4, stream);
    k_pool<<<HN_BLK, 256, 0, stream>>>(atom, batch, pooled, HN_I);
    k_head<<<N_GRAPHS, 128, 0, stream>>>(pooled, W1, b1, W2, b2, out);
}
